// GATv2Convolution_81140522156082
// MI455X (gfx1250) — compile-verified, matrix operand order checked by reading, not run on hardware
//
#include <hip/hip_runtime.h>
#include <hip/hip_bf16.h>
#include <math.h>

typedef __attribute__((ext_vector_type(16))) _Float16 v16h;
typedef __attribute__((ext_vector_type(8)))  _Float16 v8h;
typedef __attribute__((ext_vector_type(8)))  float    v8f;

#define F_IN  165
#define KP1   192      // F_IN padded to multiple of 32
#define HDIM  128
#define CDIM  2
#define NCP2  16       // layer-2 output cols padded to one 16-wide tile
#define SLOPE 0.2f
#define NEG_BIG -3.0e38f

union fragu { v16h v; v8h h[2]; };

// ---------------------------------------------------------------------------
// Operand packing: fp32 -> zero-padded fp16, branch-free GEMM fragments after.
// ---------------------------------------------------------------------------
__global__ void pack_a_f16_kernel(const float* __restrict__ A,
                                  _Float16* __restrict__ Ah,
                                  int M, int K, int KPp)
{
    size_t i = (size_t)blockIdx.x * blockDim.x + threadIdx.x;
    if (i >= (size_t)M * KPp) return;
    const int r = (int)(i / KPp);
    const int k = (int)(i % KPp);
    Ah[i] = (k < K) ? (_Float16)A[(size_t)r * K + k] : (_Float16)0.0f;
}

// B[K,Nc] row-major (f32) -> Bt[NcP, KPp] (f16), so 8 consecutive k are contiguous
__global__ void pack_bt_f16_kernel(const float* __restrict__ B,
                                   _Float16* __restrict__ Bt,
                                   int K, int KPp, int Nc, int NcP)
{
    size_t i = (size_t)blockIdx.x * blockDim.x + threadIdx.x;
    if (i >= (size_t)NcP * KPp) return;
    const int c = (int)(i / KPp);
    const int k = (int)(i % KPp);
    Bt[i] = (c < Nc && k < K) ? (_Float16)B[(size_t)k * Nc + c] : (_Float16)0.0f;
}

// ---------------------------------------------------------------------------
// WMMA GEMM:  C[M,Nc] = Ah[M,KP] @ Bt[NT*16,KP]^T   (f16 in, f32 out)
// One wave32 per 16-row stripe; NT 16-col tiles per wave so the A fragment is
// reused NT times per k-step. KP compile-time -> fully unrolled wmma chain.
// Rows assumed multiple of 16 (N = 50000 = 3125*16).
// ---------------------------------------------------------------------------
template <int KP, int NT>
__global__ void wmma_gemm_kernel(const _Float16* __restrict__ Ah,
                                 const _Float16* __restrict__ Bt,
                                 float* __restrict__ C, int Nc)
{
    const int lane  = threadIdx.x;            // 0..31
    const int l15   = lane & 15;
    const int kbase = (lane >> 4) << 3;       // 0 or 8
    const int row   = blockIdx.x * 16 + l15;

    const _Float16* arow  = Ah + (size_t)row * KP + kbase;
    const _Float16* bcol0 = Bt + (size_t)l15 * KP + kbase;

    v8f acc[NT];
#pragma unroll
    for (int nt = 0; nt < NT; ++nt) acc[nt] = (v8f){};

#pragma unroll
    for (int k0 = 0; k0 < KP; k0 += 32) {
        fragu a;
        a.h[0] = *(const v8h*)(arow + k0);        // K = kbase + k0 + 0..7
        a.h[1] = *(const v8h*)(arow + k0 + 16);   // K = kbase + k0 + 16..23
#pragma unroll
        for (int nt = 0; nt < NT; ++nt) {
            const _Float16* bc = bcol0 + (size_t)nt * 16 * KP;
            fragu b;
            b.h[0] = *(const v8h*)(bc + k0);
            b.h[1] = *(const v8h*)(bc + k0 + 16);
            acc[nt] = __builtin_amdgcn_wmma_f32_16x16x32_f16(
                false, a.v, false, b.v, (short)0, acc[nt], false, false);
        }
    }

    // C/D layout: VGPR v -> M = v + 8*(lane>=16), N = lane&15
    const int mbase = blockIdx.x * 16 + kbase;
#pragma unroll
    for (int nt = 0; nt < NT; ++nt) {
        const int col = nt * 16 + l15;
        if (col < Nc) {
#pragma unroll
            for (int v = 0; v < 8; ++v)
                C[(size_t)(mbase + v) * Nc + col] = acc[nt][v];
        }
    }
}

// ---------------------------------------------------------------------------
// Helpers
// ---------------------------------------------------------------------------
__global__ void fill_f32_kernel(float* __restrict__ p, float v, size_t n)
{
    size_t i = (size_t)blockIdx.x * blockDim.x + threadIdx.x;
    if (i < n) p[i] = v;
}

__device__ inline void atomicMaxF(float* addr, float val)
{
    unsigned int* ua  = (unsigned int*)addr;
    unsigned int  old = __float_as_uint(*addr);
    while (__uint_as_float(old) < val) {
        unsigned int assumed = old;
        old = atomicCAS(ua, assumed, __float_as_uint(val));
        if (old == assumed) break;
    }
}

__device__ inline void edge_endpoints(const int* __restrict__ ei, int E,
                                      int id, int& src, int& dst)
{
    if (id < E) { src = ei[id]; dst = ei[E + id]; }
    else        { src = dst = id - E; }   // self loop
}

// ---------------------------------------------------------------------------
// Layer-1 edge phase (128-wide): one wave32 per edge, 4 channels per lane
// ---------------------------------------------------------------------------
__global__ void edge_score_h_kernel(const float* __restrict__ XL,
                                    const float* __restrict__ XR,
                                    const float* __restrict__ att,
                                    const int* __restrict__ ei,
                                    int E, int Etot,
                                    float* __restrict__ escore,
                                    float* __restrict__ segmax)
{
    const int wid = blockIdx.x * blockDim.y + threadIdx.y;
    if (wid >= Etot) return;
    const int lane = threadIdx.x;
    int src, dst;
    edge_endpoints(ei, E, wid, src, dst);

    float part = 0.0f;
#pragma unroll
    for (int j = 0; j < 4; ++j) {
        const int c = lane * 4 + j;
        float v = XL[(size_t)src * HDIM + c] + XR[(size_t)dst * HDIM + c];
        v = (v > 0.0f) ? v : SLOPE * v;
        part += v * att[c];
    }
#pragma unroll
    for (int o = 16; o > 0; o >>= 1) part += __shfl_xor(part, o, 32);

    if (lane == 0) {
        escore[wid] = part;
        atomicMaxF(&segmax[dst], part);
    }
}

// shared by both layers: p = exp(e - max[dst]); denom[dst] += p
__global__ void edge_expsum_kernel(const float* __restrict__ escore,
                                   const int* __restrict__ ei,
                                   int E, int Etot,
                                   const float* __restrict__ segmax,
                                   float* __restrict__ p,
                                   float* __restrict__ denom)
{
    const int i = blockIdx.x * blockDim.x + threadIdx.x;
    if (i >= Etot) return;
    const int dst = (i < E) ? ei[E + i] : (i - E);
    const float pv = expf(escore[i] - segmax[dst]);
    p[i] = pv;
    atomicAdd(&denom[dst], pv);
}

__global__ void edge_aggregate_h_kernel(const float* __restrict__ XL,
                                        const int* __restrict__ ei,
                                        int E, int Etot,
                                        const float* __restrict__ p,
                                        const float* __restrict__ denom,
                                        float* __restrict__ agg)
{
    const int wid = blockIdx.x * blockDim.y + threadIdx.y;
    if (wid >= Etot) return;
    const int lane = threadIdx.x;
    int src, dst;
    edge_endpoints(ei, E, wid, src, dst);
    const float alpha = p[wid] / denom[dst];
#pragma unroll
    for (int j = 0; j < 4; ++j) {
        const int c = lane * 4 + j;
        atomicAdd(&agg[(size_t)dst * HDIM + c],
                  alpha * XL[(size_t)src * HDIM + c]);
    }
}

// h = elu(agg + b1 + xlin + blin1), emitted directly as f16 (feeds only GEMMs)
__global__ void combine_elu_kernel(const float* __restrict__ agg,
                                   const float* __restrict__ b1,
                                   const float* __restrict__ xlin,
                                   const float* __restrict__ blin1,
                                   _Float16* __restrict__ h16, int total)
{
    const int i = blockIdx.x * blockDim.x + threadIdx.x;
    if (i >= total) return;
    const int c = i & (HDIM - 1);
    const float v = agg[i] + b1[c] + xlin[i] + blin1[c];
    h16[i] = (_Float16)((v > 0.0f) ? v : (expf(v) - 1.0f));
}

// ---------------------------------------------------------------------------
// Layer-2 edge phase (C=2): one thread per edge
// ---------------------------------------------------------------------------
__global__ void edge_score_c_kernel(const float* __restrict__ HL,
                                    const float* __restrict__ HR,
                                    const float* __restrict__ att,
                                    const int* __restrict__ ei,
                                    int E, int Etot,
                                    float* __restrict__ escore,
                                    float* __restrict__ segmax)
{
    const int i = blockIdx.x * blockDim.x + threadIdx.x;
    if (i >= Etot) return;
    int src, dst;
    edge_endpoints(ei, E, i, src, dst);
    float e = 0.0f;
#pragma unroll
    for (int c = 0; c < CDIM; ++c) {
        float v = HL[(size_t)src * CDIM + c] + HR[(size_t)dst * CDIM + c];
        v = (v > 0.0f) ? v : SLOPE * v;
        e += v * att[c];
    }
    escore[i] = e;
    atomicMaxF(&segmax[dst], e);
}

__global__ void edge_aggregate_c_kernel(const float* __restrict__ HL,
                                        const int* __restrict__ ei,
                                        int E, int Etot,
                                        const float* __restrict__ p,
                                        const float* __restrict__ denom,
                                        float* __restrict__ agg)
{
    const int i = blockIdx.x * blockDim.x + threadIdx.x;
    if (i >= Etot) return;
    int src, dst;
    edge_endpoints(ei, E, i, src, dst);
    const float alpha = p[i] / denom[dst];
#pragma unroll
    for (int c = 0; c < CDIM; ++c)
        atomicAdd(&agg[(size_t)dst * CDIM + c],
                  alpha * HL[(size_t)src * CDIM + c]);
}

// out = log_softmax(agg2 + b2 + hlin + blin2) over C=2
__global__ void final_logsoftmax_kernel(const float* __restrict__ agg,
                                        const float* __restrict__ b2,
                                        const float* __restrict__ hlin,
                                        const float* __restrict__ blin2,
                                        float* __restrict__ out, int Nn)
{
    const int n = blockIdx.x * blockDim.x + threadIdx.x;
    if (n >= Nn) return;
    float o[CDIM];
#pragma unroll
    for (int c = 0; c < CDIM; ++c)
        o[c] = agg[(size_t)n * CDIM + c] + b2[c]
             + hlin[(size_t)n * CDIM + c] + blin2[c];
    float mx = o[0];
#pragma unroll
    for (int c = 1; c < CDIM; ++c) mx = fmaxf(mx, o[c]);
    float s = 0.0f;
#pragma unroll
    for (int c = 0; c < CDIM; ++c) s += expf(o[c] - mx);
    const float ls = mx + logf(s);
#pragma unroll
    for (int c = 0; c < CDIM; ++c)
        out[(size_t)n * CDIM + c] = o[c] - ls;
}

// second tuple element: edge_index echoed into d_out (as float values)
__global__ void copy_edges_kernel(const int* __restrict__ ei,
                                  float* __restrict__ out, int n)
{
    const int i = blockIdx.x * blockDim.x + threadIdx.x;
    if (i < n) out[i] = (float)ei[i];
}

// ---------------------------------------------------------------------------
// Launcher
// ---------------------------------------------------------------------------
extern "C" void kernel_launch(void* const* d_in, const int* in_sizes, int n_in,
                              void* d_out, int out_size, void* d_ws, size_t ws_size,
                              hipStream_t stream)
{
    const float* x     = (const float*)d_in[0];
    const float* W1l   = (const float*)d_in[1];
    const float* W1r   = (const float*)d_in[2];
    const float* a1    = (const float*)d_in[3];
    const float* b1    = (const float*)d_in[4];
    const float* Wlin1 = (const float*)d_in[5];
    const float* blin1 = (const float*)d_in[6];
    const float* W2l   = (const float*)d_in[7];
    const float* W2r   = (const float*)d_in[8];
    const float* a2    = (const float*)d_in[9];
    const float* b2    = (const float*)d_in[10];
    const float* Wlin2 = (const float*)d_in[11];
    const float* blin2 = (const float*)d_in[12];
    const int*   ei    = (const int*)d_in[13];

    const int N    = in_sizes[0] / F_IN;
    const int E    = in_sizes[13] / 2;
    const int Etot = E + N;

    // ---- workspace carve-up (16-byte aligned slices) ----
    char*  ws  = (char*)d_ws;
    size_t off = 0;
    auto carve = [&](size_t bytes) {
        void* p = ws + off;
        off += (bytes + 15) & ~(size_t)15;
        return p;
    };
    float*    XL    = (float*)carve((size_t)N * HDIM * 4);
    float*    XR    = (float*)carve((size_t)N * HDIM * 4);
    float*    XLIN  = (float*)carve((size_t)N * HDIM * 4);
    float*    AGG1  = (float*)carve((size_t)N * HDIM * 4);
    _Float16* Hh    = (_Float16*)carve((size_t)N * HDIM * 2);
    _Float16* Xh    = (_Float16*)carve((size_t)N * KP1 * 2);
    _Float16* Bt1l  = (_Float16*)carve((size_t)HDIM * KP1 * 2);
    _Float16* Bt1r  = (_Float16*)carve((size_t)HDIM * KP1 * 2);
    _Float16* Bt1s  = (_Float16*)carve((size_t)HDIM * KP1 * 2);
    _Float16* Bt2l  = (_Float16*)carve((size_t)NCP2 * HDIM * 2);
    _Float16* Bt2r  = (_Float16*)carve((size_t)NCP2 * HDIM * 2);
    _Float16* Bt2s  = (_Float16*)carve((size_t)NCP2 * HDIM * 2);
    float*    E1    = (float*)carve((size_t)Etot * 4);
    float*    P1    = (float*)carve((size_t)Etot * 4);
    float*    M1    = (float*)carve((size_t)N * 4);
    float*    D1    = (float*)carve((size_t)N * 4);
    float*    HL2   = (float*)carve((size_t)N * CDIM * 4);
    float*    HR2   = (float*)carve((size_t)N * CDIM * 4);
    float*    HLIN2 = (float*)carve((size_t)N * CDIM * 4);
    float*    E2    = (float*)carve((size_t)Etot * 4);
    float*    P2    = (float*)carve((size_t)Etot * 4);
    float*    M2    = (float*)carve((size_t)N * 4);
    float*    D2    = (float*)carve((size_t)N * 4);
    float*    AGG2  = (float*)carve((size_t)N * CDIM * 4);
    (void)ws_size; (void)n_in; (void)out_size;

    float* out = (float*)d_out;

    const int T   = 256;
    const int mt1 = (N + 15) / 16;       // 3125 row tiles (exact)
    const dim3 gW(32, 1, 1);             // one wave per block

    // ---- pack operands to padded f16 ----
    {
        size_t na = (size_t)N * KP1;
        pack_a_f16_kernel<<<(int)((na + T - 1) / T), T, 0, stream>>>(x, Xh, N, F_IN, KP1);
        size_t nb1 = (size_t)HDIM * KP1;
        int gb1 = (int)((nb1 + T - 1) / T);
        pack_bt_f16_kernel<<<gb1, T, 0, stream>>>(W1l,   Bt1l, F_IN, KP1, HDIM, HDIM);
        pack_bt_f16_kernel<<<gb1, T, 0, stream>>>(W1r,   Bt1r, F_IN, KP1, HDIM, HDIM);
        pack_bt_f16_kernel<<<gb1, T, 0, stream>>>(Wlin1, Bt1s, F_IN, KP1, HDIM, HDIM);
        size_t nb2 = (size_t)NCP2 * HDIM;
        int gb2 = (int)((nb2 + T - 1) / T);
        pack_bt_f16_kernel<<<gb2, T, 0, stream>>>(W2l,   Bt2l, HDIM, HDIM, CDIM, NCP2);
        pack_bt_f16_kernel<<<gb2, T, 0, stream>>>(W2r,   Bt2r, HDIM, HDIM, CDIM, NCP2);
        pack_bt_f16_kernel<<<gb2, T, 0, stream>>>(Wlin2, Bt2s, HDIM, HDIM, CDIM, NCP2);
    }

    // ---- layer 1 dense transforms via WMMA (16 rows x 128 cols per wave) ----
    wmma_gemm_kernel<KP1, HDIM / 16><<<mt1, gW, 0, stream>>>(Xh, Bt1l, XL,   HDIM);
    wmma_gemm_kernel<KP1, HDIM / 16><<<mt1, gW, 0, stream>>>(Xh, Bt1r, XR,   HDIM);
    wmma_gemm_kernel<KP1, HDIM / 16><<<mt1, gW, 0, stream>>>(Xh, Bt1s, XLIN, HDIM);

    // ---- init segment buffers ----
    {
        fill_f32_kernel<<<(N + T - 1) / T, T, 0, stream>>>(M1, NEG_BIG, (size_t)N);
        fill_f32_kernel<<<(N + T - 1) / T, T, 0, stream>>>(D1, 0.0f, (size_t)N);
        size_t nh = (size_t)N * HDIM;
        fill_f32_kernel<<<(int)((nh + T - 1) / T), T, 0, stream>>>(AGG1, 0.0f, nh);
        fill_f32_kernel<<<(N + T - 1) / T, T, 0, stream>>>(M2, NEG_BIG, (size_t)N);
        fill_f32_kernel<<<(N + T - 1) / T, T, 0, stream>>>(D2, 0.0f, (size_t)N);
        size_t nc = (size_t)N * CDIM;
        fill_f32_kernel<<<(int)((nc + T - 1) / T), T, 0, stream>>>(AGG2, 0.0f, nc);
    }

    // ---- layer 1 edge phase ----
    {
        dim3 blk(32, 8);
        int  grd = (Etot + 7) / 8;
        edge_score_h_kernel<<<grd, blk, 0, stream>>>(XL, XR, a1, ei, E, Etot, E1, M1);
        edge_expsum_kernel<<<(Etot + T - 1) / T, T, 0, stream>>>(E1, ei, E, Etot, M1, P1, D1);
        edge_aggregate_h_kernel<<<grd, blk, 0, stream>>>(XL, ei, E, Etot, P1, D1, AGG1);
        size_t nh = (size_t)N * HDIM;
        combine_elu_kernel<<<(int)((nh + T - 1) / T), T, 0, stream>>>(
            AGG1, b1, XLIN, blin1, Hh, (int)nh);
    }

    // ---- layer 2 dense transforms via WMMA (K=128, one padded 16-col tile) ----
    wmma_gemm_kernel<HDIM, 1><<<mt1, gW, 0, stream>>>(Hh, Bt2l, HL2,   CDIM);
    wmma_gemm_kernel<HDIM, 1><<<mt1, gW, 0, stream>>>(Hh, Bt2r, HR2,   CDIM);
    wmma_gemm_kernel<HDIM, 1><<<mt1, gW, 0, stream>>>(Hh, Bt2s, HLIN2, CDIM);

    // ---- layer 2 edge phase + output ----
    {
        edge_score_c_kernel<<<(Etot + T - 1) / T, T, 0, stream>>>(
            HL2, HR2, a2, ei, E, Etot, E2, M2);
        edge_expsum_kernel<<<(Etot + T - 1) / T, T, 0, stream>>>(E2, ei, E, Etot, M2, P2, D2);
        edge_aggregate_c_kernel<<<(Etot + T - 1) / T, T, 0, stream>>>(
            HL2, ei, E, Etot, P2, D2, AGG2);
        final_logsoftmax_kernel<<<(N + T - 1) / T, T, 0, stream>>>(
            AGG2, b2, HLIN2, blin2, out, N);
        copy_edges_kernel<<<(2 * E + T - 1) / T, T, 0, stream>>>(
            ei, out + (size_t)N * CDIM, 2 * E);
    }
}
